// DeepTourModel_59854664237655
// MI455X (gfx1250) — matrix operand order repaired and checked by reading, not compiled, not run on hardware
//
#include <hip/hip_runtime.h>

// ---------------------------------------------------------------------------
// DeepTour heterogeneous-GNN forward for MI455X (gfx1250, wave32, WMMA).
// - All dense projections: v_wmma_f32_16x16x32_bf16, LDS-staged tiles.
// - Attention algebraically factored (dot(X@W,q) == X.(W@q)): kills the
//   440MB xh intermediate and 90% of attention FLOPs.
// - GEMM-only intermediates (spot_in, src_all, tgt, spottgt, wordtgt) are
//   kept in bf16: ~300MB less HBM traffic on a 23.3TB/s-bound pipeline.
// - bf16 A tiles staged with global_load_async_to_lds_b128 (ASYNCcnt path);
//   fp32 operands staged via b128 loads + packed b32 LDS stores.
// Workspace requirement: ~380 MB (phase-aliased scratch region).
// ---------------------------------------------------------------------------

typedef __attribute__((ext_vector_type(16))) __bf16    v16bf;
typedef __attribute__((ext_vector_type(8)))  float     v8f;
typedef __attribute__((ext_vector_type(8)))  unsigned  v8u;

static constexpr long long cN    = 42852;    // spots
static constexpr long long cW    = 30000;    // words
static constexpr long long cE    = 257112;   // spot-spot edges
static constexpr long long cEW   = 400000;   // word->spot edges
static constexpr long long cESW  = 400000;   // spot->word edges
static constexpr long long cKSEL = 128556;   // sampled edges
static constexpr long long cME   = cKSEL + cN; // sampled + self loops
static constexpr long long cND   = 2142;     // dist embeddings

static constexpr int LSTR = 20;              // LDS tile row stride in dwords
                                             // (80B: 16B-aligned for async
                                             // b128, conflict-free: 20*m%64
                                             // distinct for m=0..15)

#define DEV static __device__ __forceinline__

DEV unsigned short f2bf(float f) {
  unsigned u = __float_as_uint(f);
  u += 0x7FFFu + ((u >> 16) & 1u);           // round-to-nearest-even
  return (unsigned short)(u >> 16);
}
DEV unsigned pack2(float lo, float hi) {     // two bf16 in one dword
  return (unsigned)f2bf(lo) | ((unsigned)f2bf(hi) << 16);
}
DEV float lrelu(float x) { return x > 0.f ? x : 0.2f * x; }
DEV float sigmoidf(float x) { return 1.f / (1.f + __expf(-x)); }
DEV float wred(float v) {                     // wave32 butterfly all-reduce
#pragma unroll
  for (int o = 16; o > 0; o >>= 1) v += __shfl_xor(v, o, 32);
  return v;
}
DEV float gumbel_hash(unsigned i) {
  unsigned x = i * 0x9E3779B9u + 0x85EBCA6Bu;
  x ^= x >> 16; x *= 0x7FEB352Du; x ^= x >> 15; x *= 0x846CA68Bu; x ^= x >> 16;
  const float u = ((x >> 8) + 0.5f) * (1.0f / 16777216.0f);   // (0,1)
  return -__logf(-__logf(u));
}
DEV void atomicMaxF(float* addr, float val) {
  int* ia = (int*)addr;
  int old = __float_as_int(*addr);
  while (__int_as_float(old) < val) {
    int prev = atomicCAS(ia, old, __float_as_int(val));
    if (prev == old) break;
    old = prev;
  }
}

// ---------------------------------------------------------------------------
// Fragment gather + 4 chained WMMAs for one 16x64 strip of the block tile.
// A frag: K pairs per dword, half-wave selects K+16.
// B frag: K pairs per dword, half-wave selects K+16.
// ---------------------------------------------------------------------------
DEV void wmma_step(const unsigned* __restrict__ AsU, const unsigned* __restrict__ BsU,
                   int wv, int m, int half, v8f acc[4])
{
  union { v8u u; v16bf v; } af;
#pragma unroll
  for (int v = 0; v < 8; ++v)
    af.u[v] = AsU[(wv * 16 + m) * LSTR + v + (half << 3)];
#pragma unroll
  for (int nt = 0; nt < 4; ++nt) {
    union { v8u u; v16bf v; } bf;
#pragma unroll
    for (int v = 0; v < 8; ++v)
      bf.u[v] = BsU[(nt * 16 + m) * LSTR + v + (half << 3)];
    acc[nt] = __builtin_amdgcn_wmma_f32_16x16x32_bf16(
        false, af.v, false, bf.v, (short)0, acc[nt], false, false);
  }
}

// ---------------------------------------------------------------------------
// Generic tiled WMMA GEMM:  C[M,Nc] (+)= A[M,K] @ B  (+ bias)
//   ABF : A is bf16 (row-major, lda) else fp32
//   BT  : B stored [Nc,K] (row-major, ldb=K-stride)  else  [K,Nc]
//   ACC : C += result (fp32 C only)   BIAS: add bias[col]
//   OBF : C stored as bf16 (else fp32)
// Block = 128 threads (4 waves) -> 64x64 tile, K staged in 32-wide bf16 LDS
// tiles. bf16 A staged via global_load_async_to_lds_b128 (ASYNCcnt drained
// with s_wait_asynccnt before the barrier); fp32 operands staged via b128
// loads + packed b32 LDS stores. Guards exist only in M-tail rows and the
// K-tail tile (only K=300 takes it).
// ---------------------------------------------------------------------------
template<bool ABF, bool BT, bool ACC, bool BIAS, bool OBF>
__global__ __launch_bounds__(128)
void gemm_wmma_k(const void* __restrict__ Ap, const float* __restrict__ B,
                 const float* __restrict__ bias, void* __restrict__ Cv,
                 int M, int Nc, int K, int lda, int ldb, int ldc)
{
  __shared__ unsigned short As[64][2 * LSTR];   // [row][k], 80B row stride
  __shared__ unsigned short Bs[64][2 * LSTR];   // [col][k] (B staged transposed)
  const int tid = threadIdx.x, lane = tid & 31, wv = tid >> 5;
  const int row0 = blockIdx.x * 64, col0 = blockIdx.y * 64;
  const int m = lane & 15, half = lane >> 4;
  const unsigned* AsU = (const unsigned*)&As[0][0];
  const unsigned* BsU = (const unsigned*)&Bs[0][0];

  v8f acc[4];
#pragma unroll
  for (int i = 0; i < 4; ++i)
#pragma unroll
    for (int j = 0; j < 8; ++j) acc[i][j] = 0.f;

  // staging coordinates
  const int rA  = tid >> 1, kbA = (tid & 1) * 16;     // A / BT-B: row-mapped, 16 contiguous k
  const int kB  = tid >> 2, nbB = (tid & 3) * 16;     // non-BT B: k-mapped, 16 contiguous n
  const long long grA = row0 + rA;
  const long long gnA = col0 + rA;
  const bool rowOK  = grA < M;
  const bool bColOK = gnA < Nc;
  const bool bFast  = (col0 + nbB + 15) < Nc;         // non-BT fast-path col guard
  unsigned* AsRow = (unsigned*)&As[rA][0] + (kbA >> 1);
  unsigned* BsRow = (unsigned*)&Bs[rA][0] + (kbA >> 1);
  const unsigned ldsA = (unsigned)(size_t)AsRow;      // LDS byte offset (low 32b)

  const int Kmain = K & ~31;
  int kk = 0;
  for (; kk < Kmain; kk += 32) {
    // ---- stage A ----
    if (ABF) {
      if (rowOK) {
        // async DMA: each lane moves 2x16B straight into LDS (ASYNCcnt).
        const unsigned short* gp = (const unsigned short*)Ap +
                                   grA * (long long)lda + kk + kbA;
        asm volatile(
            "global_load_async_to_lds_b128 %0, %1, off\n"
            "global_load_async_to_lds_b128 %0, %1, off offset:16"
            :: "v"(ldsA), "v"(gp) : "memory");
      } else {
#pragma unroll
        for (int q = 0; q < 8; ++q) AsRow[q] = 0u;
      }
    } else {
      if (rowOK) {
        const float4* s4 = (const float4*)((const float*)Ap +
                                           grA * (long long)lda + kk + kbA);
#pragma unroll
        for (int q = 0; q < 4; ++q) {
          const float4 f = s4[q];
          AsRow[2 * q]     = pack2(f.x, f.y);
          AsRow[2 * q + 1] = pack2(f.z, f.w);
        }
      } else {
#pragma unroll
        for (int q = 0; q < 8; ++q) AsRow[q] = 0u;
      }
    }
    // ---- stage B ----
    if (BT) {                        // B stored [Nc,K]: contiguous along k
      if (bColOK) {
        const float4* s4 = (const float4*)(B + gnA * (long long)ldb + kk + kbA);
#pragma unroll
        for (int q = 0; q < 4; ++q) {
          const float4 f = s4[q];
          BsRow[2 * q]     = pack2(f.x, f.y);
          BsRow[2 * q + 1] = pack2(f.z, f.w);
        }
      } else {
#pragma unroll
        for (int q = 0; q < 8; ++q) BsRow[q] = 0u;
      }
    } else {                         // B stored [K,Nc]: contiguous along n
      if (bFast) {
        const float4* s4 = (const float4*)(B + (long long)(kk + kB) * ldb + col0 + nbB);
#pragma unroll
        for (int q = 0; q < 4; ++q) {
          const float4 f = s4[q];
          Bs[nbB + 4 * q + 0][kB] = f2bf(f.x);
          Bs[nbB + 4 * q + 1][kB] = f2bf(f.y);
          Bs[nbB + 4 * q + 2][kB] = f2bf(f.z);
          Bs[nbB + 4 * q + 3][kB] = f2bf(f.w);
        }
      } else {
#pragma unroll
        for (int j = 0; j < 16; ++j) {
          const int n = nbB + j;
          unsigned short h = 0;
          if (col0 + n < Nc) h = f2bf(B[(long long)(kk + kB) * ldb + col0 + n]);
          Bs[n][kB] = h;
        }
      }
    }
    if (ABF) asm volatile("s_wait_asynccnt 0x0" ::: "memory");
    __syncthreads();
    wmma_step(AsU, BsU, wv, m, half, acc);
    __syncthreads();
  }

  if (kk < K) {                      // guarded tail tile (only K=300 here)
#pragma unroll
    for (int j = 0; j < 16; ++j) {
      const int k = kbA + j; const long long gk = kk + k;
      unsigned short h = 0;
      if (rowOK && gk < K) {
        if (ABF) h = ((const unsigned short*)Ap)[grA * (long long)lda + gk];
        else     h = f2bf(((const float*)Ap)[grA * (long long)lda + gk]);
      }
      As[rA][k] = h;
    }
    if (BT) {
#pragma unroll
      for (int j = 0; j < 16; ++j) {
        const int k = kbA + j; const long long gk = kk + k;
        unsigned short h = 0;
        if (bColOK && gk < K) h = f2bf(B[gnA * (long long)ldb + gk]);
        Bs[rA][k] = h;
      }
    } else {
      const long long gk = kk + kB;
#pragma unroll
      for (int j = 0; j < 16; ++j) {
        const int n = nbB + j;
        unsigned short h = 0;
        if (gk < K && col0 + n < Nc) h = f2bf(B[gk * (long long)ldb + col0 + n]);
        Bs[n][kB] = h;
      }
    }
    __syncthreads();
    wmma_step(AsU, BsU, wv, m, half, acc);
    __syncthreads();
  }

  // D layout: lane(n=l&15), VGPR i -> row i + 8*(l>>4)
#pragma unroll
  for (int nt = 0; nt < 4; ++nt) {
    const int col = col0 + nt * 16 + m;
    if (col >= Nc) continue;
#pragma unroll
    for (int i = 0; i < 8; ++i) {
      const int row = row0 + wv * 16 + i + half * 8;
      if (row < M) {
        float v = acc[nt][i];
        if (BIAS) v += bias[col];
        const long long idx = (long long)row * ldc + col;
        if (OBF) {
          ((unsigned short*)Cv)[idx] = f2bf(v);
        } else {
          float* Cf = (float*)Cv;
          if (ACC) v += Cf[idx];
          Cf[idx] = v;
        }
      }
    }
  }
}

template<bool ABF, bool BT, bool ACC, bool BIAS, bool OBF>
static void gemm(hipStream_t s, const void* A, const float* B, const float* bias,
                 void* C, int M, int Nc, int K, int lda, int ldb, int ldc)
{
  dim3 g((unsigned)((M + 63) / 64), (unsigned)((Nc + 63) / 64)), b(128);
  gemm_wmma_k<ABF, BT, ACC, BIAS, OBF><<<g, b, 0, s>>>(A, B, bias, C, M, Nc, K, lda, ldb, ldc);
}

// ---------------------------------------------------------------------------
// Attention: wq[h] = W_h @ q_h   (4*512 outputs, 128-dot each)
__global__ __launch_bounds__(256)
void wq_k(const float* __restrict__ W, const float* __restrict__ q, float* __restrict__ wq)
{
  const int t = blockIdx.x * 256 + threadIdx.x;
  if (t >= 2048) return;
  const int h = t >> 9, f = t & 511;
  const float* Wp = W + ((long long)h * 512 + f) * 128;
  const float* qp = q + h * 128;
  float a = 0.f;
  for (int d = 0; d < 128; ++d) a += Wp[d] * qp[d];
  wq[t] = a;
}

// scores = lrelu(Xs . wq_h), softmax over 5 splits, combined input in bf16.
// One wave per (node, head); each lane owns 16 contiguous features.
__global__ __launch_bounds__(256)
void attn_combine_k(const float* __restrict__ x, const float* __restrict__ wq,
                    unsigned short* __restrict__ xbar)
{
  const long long gw = ((long long)blockIdx.x * 256 + threadIdx.x) >> 5;
  const int lane = threadIdx.x & 31;
  const long long n = gw >> 2; const int h = (int)(gw & 3);
  if (n >= cN) return;
  const float* xp = x + n * 2560;
  const float* wp = wq + h * 512 + lane * 16;
  float xr[5][16], sc[5];
#pragma unroll
  for (int s2 = 0; s2 < 5; ++s2) {
    const float4* s4 = (const float4*)(xp + s2 * 512 + lane * 16);
    float a = 0.f;
#pragma unroll
    for (int q = 0; q < 4; ++q) {
      const float4 f = s4[q];
      xr[s2][4 * q + 0] = f.x; xr[s2][4 * q + 1] = f.y;
      xr[s2][4 * q + 2] = f.z; xr[s2][4 * q + 3] = f.w;
      a += f.x * wp[4 * q] + f.y * wp[4 * q + 1] + f.z * wp[4 * q + 2] + f.w * wp[4 * q + 3];
    }
    sc[s2] = wred(a);
  }
  float mx = -3.0e38f;
#pragma unroll
  for (int s2 = 0; s2 < 5; ++s2) { sc[s2] = lrelu(sc[s2]); mx = fmaxf(mx, sc[s2]); }
  float se = 0.f;
#pragma unroll
  for (int s2 = 0; s2 < 5; ++s2) { sc[s2] = __expf(sc[s2] - mx); se += sc[s2]; }
  const float inv = 1.f / se;
  unsigned* xb = (unsigned*)(xbar + ((long long)h * cN + n) * 512) + lane * 8;
#pragma unroll
  for (int j = 0; j < 8; ++j) {
    float a0 = 0.f, a1 = 0.f;
#pragma unroll
    for (int s2 = 0; s2 < 5; ++s2) {
      a0 += sc[s2] * xr[s2][2 * j];
      a1 += sc[s2] * xr[s2][2 * j + 1];
    }
    xb[j] = pack2(a0 * inv, a1 * inv);
  }
}

// ---------------------------------------------------------------------------
// Edge scoring: key = -||q - v|| + Gumbel(hash(e)), stored as sortable u32.
__global__ __launch_bounds__(256)
void edge_logit_k(const int* __restrict__ v0, const int* __restrict__ v1,
                  const int* __restrict__ dist, const float* __restrict__ qn,
                  const float* __restrict__ qd, const float* __restrict__ vn,
                  unsigned* __restrict__ keyu)
{
  const long long e = ((long long)blockIdx.x * 256 + threadIdx.x) >> 5;
  const int lane = threadIdx.x & 31;
  if (e >= cE) return;
  const long long s = v0[e], d = v1[e];
  const long long di = (dist[e] / 20) / 20;
  float a = 0.f;
#pragma unroll
  for (int j = 0; j < 4; ++j) {
    const int c = lane + j * 32;
    const float q = qn[s * 128 + c] + qd[di * 128 + c];
    const float t = q - vn[d * 128 + c];
    a += t * t;
  }
  a = wred(a);
  if (lane == 0) {
    const float key = -sqrtf(a) + gumbel_hash((unsigned)e);
    const unsigned u = __float_as_uint(key);
    keyu[e] = (u & 0x80000000u) ? ~u : (u | 0x80000000u);
  }
}

// Device-side binary search for the K_SEL-th largest sortable key.
// state: [0]=lo [1]=hi [2]=mid [3]=count [4]=thr [5]=nsel
__global__ void bisect_init_k(unsigned* st)
{
  if (threadIdx.x == 0) {
    st[0] = 0u; st[1] = 0xFFFFFFFFu; st[2] = 0x80000000u;
    st[3] = 0u; st[4] = 0u; st[5] = 0u;
  }
}
__global__ __launch_bounds__(256)
void count_ge_k(const unsigned* __restrict__ keyu, unsigned* __restrict__ st)
{
  const long long i = (long long)blockIdx.x * 256 + threadIdx.x;
  const unsigned mid = st[2];
  const bool p = (i < cE) && (keyu[i] >= mid);
  const unsigned long long b = __ballot(p);
  if ((threadIdx.x & 31) == 0 && b)
    atomicAdd(&st[3], (unsigned)__popcll(b));
}
__global__ void bisect_step_k(unsigned* st)
{
  if (threadIdx.x != 0) return;
  unsigned lo = st[0], hi = st[1];
  const unsigned mid = st[2], c = st[3];
  if (c >= (unsigned)cKSEL) lo = mid; else hi = mid - 1u;
  st[0] = lo; st[1] = hi; st[3] = 0u;
  st[2] = (unsigned)(((unsigned long long)lo + hi + 1ull) >> 1);
  st[4] = lo;      // threshold after final iteration
}
__global__ __launch_bounds__(256)
void select_gt_k(const unsigned* __restrict__ keyu, const int* __restrict__ v0,
                 const int* __restrict__ v1, unsigned* __restrict__ st,
                 int* __restrict__ ss, int* __restrict__ sd, int* __restrict__ kp)
{
  const long long i = (long long)blockIdx.x * 256 + threadIdx.x;
  if (i >= cE) return;
  if (keyu[i] > st[4]) {
    const unsigned p = atomicAdd(&st[5], 1u);
    if (p < (unsigned)cKSEL) { ss[p] = v0[i]; sd[p] = v1[i]; kp[p] = (v0[i] != v1[i]); }
  }
}
__global__ __launch_bounds__(256)
void select_eq_k(const unsigned* __restrict__ keyu, const int* __restrict__ v0,
                 const int* __restrict__ v1, unsigned* __restrict__ st,
                 int* __restrict__ ss, int* __restrict__ sd, int* __restrict__ kp)
{
  const long long i = (long long)blockIdx.x * 256 + threadIdx.x;
  if (i >= cE) return;
  if (keyu[i] == st[4]) {
    const unsigned p = atomicAdd(&st[5], 1u);
    if (p < (unsigned)cKSEL) { ss[p] = v0[i]; sd[p] = v1[i]; kp[p] = (v0[i] != v1[i]); }
  }
}

// ---------------------------------------------------------------------------
// GAT: init out=bias, m=-inf, wsum=0, self-loop edges.
__global__ __launch_bounds__(256)
void gat_init_k(const float* __restrict__ gb, float* __restrict__ gout,
                float* __restrict__ m, float* __restrict__ wsum,
                int* __restrict__ ss, int* __restrict__ sd, int* __restrict__ kp)
{
  const long long t = (long long)blockIdx.x * 256 + threadIdx.x;
  if (t >= cN * 128) return;
  const long long n = t >> 7; const int d = (int)(t & 127);
  gout[t] = gb[d];
  if (d == 0) {
    m[n] = -3.0e38f; wsum[n] = 0.f;
    ss[cKSEL + n] = (int)n; sd[cKSEL + n] = (int)n; kp[cKSEL + n] = 1;
  }
}
__global__ __launch_bounds__(256)
void asad_k(const float* __restrict__ h, const float* __restrict__ as,
            const float* __restrict__ ad, float* __restrict__ out)
{
  const long long n = ((long long)blockIdx.x * 256 + threadIdx.x) >> 5;
  const int lane = threadIdx.x & 31;
  if (n >= cN) return;
  float s1 = 0.f, s2 = 0.f;
#pragma unroll
  for (int j = 0; j < 4; ++j) {
    const int c = lane + j * 32;
    const float v = h[n * 128 + c];
    s1 += v * as[c]; s2 += v * ad[c];
  }
  s1 = wred(s1); s2 = wred(s2);
  if (lane == 0) { out[2 * n] = s1; out[2 * n + 1] = s2; }
}
__global__ __launch_bounds__(256)
void gat_e_k(const int* __restrict__ ss, const int* __restrict__ sd,
             const int* __restrict__ kp, const float* __restrict__ asad,
             float* __restrict__ ebuf, float* __restrict__ mbuf)
{
  const long long i = (long long)blockIdx.x * 256 + threadIdx.x;
  if (i >= cME) return;
  const float e = kp[i] ? lrelu(asad[2 * (long long)ss[i]] + asad[2 * (long long)sd[i] + 1])
                        : -1e9f;
  ebuf[i] = e;
  atomicMaxF(&mbuf[sd[i]], e);
}
__global__ __launch_bounds__(256)
void gat_w_k(const int* __restrict__ sd, const float* __restrict__ ebuf,
             const float* __restrict__ mbuf, float* __restrict__ wbuf,
             float* __restrict__ wsum)
{
  const long long i = (long long)blockIdx.x * 256 + threadIdx.x;
  if (i >= cME) return;
  const float w = __expf(ebuf[i] - mbuf[sd[i]]);
  wbuf[i] = w;
  atomicAdd(&wsum[sd[i]], w);
}
__global__ __launch_bounds__(256)
void gat_scatter_k(const int* __restrict__ ss, const int* __restrict__ sd,
                   const float* __restrict__ wbuf, const float* __restrict__ wsum,
                   const float* __restrict__ h, float* __restrict__ gout)
{
  const long long i = ((long long)blockIdx.x * 256 + threadIdx.x) >> 5;
  const int lane = threadIdx.x & 31;
  if (i >= cME) return;
  const long long s = ss[i], d = sd[i];
  const float coeff = wbuf[i] / fmaxf(wsum[d], 1e-20f);
#pragma unroll
  for (int j = 0; j < 4; ++j) {
    const int c = lane + j * 32;
    atomicAdd(&gout[d * 128 + c], h[s * 128 + c] * coeff);
  }
}

// ---------------------------------------------------------------------------
// Bipartite scatter-mean, GRU, misc.
__global__ __launch_bounds__(256)
void fill_k(float* __restrict__ p, float v, long long n)
{
  const long long i = (long long)blockIdx.x * 256 + threadIdx.x;
  if (i < n) p[i] = v;
}
__global__ __launch_bounds__(256)
void scatter_mean_k(const int* __restrict__ src, const int* __restrict__ dst,
                    const float* __restrict__ feat, float* __restrict__ agg,
                    float* __restrict__ cnt, long long nE)
{
  const long long e = ((long long)blockIdx.x * 256 + threadIdx.x) >> 5;
  const int lane = threadIdx.x & 31;
  if (e >= nE) return;
  const long long s = src[e], d = dst[e];
#pragma unroll
  for (int j = 0; j < 4; ++j) {
    const int c = lane + j * 32;
    atomicAdd(&agg[d * 128 + c], feat[s * 128 + c]);
  }
  if (lane == 0) atomicAdd(&cnt[d], 1.f);
}
__global__ __launch_bounds__(256)
void div_cnt_k(float* __restrict__ agg, const float* __restrict__ cnt, long long rows)
{
  const long long t = (long long)blockIdx.x * 256 + threadIdx.x;
  if (t >= rows * 128) return;
  agg[t] /= fmaxf(cnt[t >> 7], 1.f);
}
__global__ __launch_bounds__(256)
void gru_k(const float* __restrict__ gi, const float* __restrict__ gh,
           const float* __restrict__ hprev, float* __restrict__ out, long long rows)
{
  const long long t = (long long)blockIdx.x * 256 + threadIdx.x;
  if (t >= rows * 128) return;
  const long long n = t >> 7; const int d = (int)(t & 127);
  const float ir = gi[n * 384 + d],        hr = gh[n * 384 + d];
  const float iz = gi[n * 384 + 128 + d],  hz = gh[n * 384 + 128 + d];
  const float in2 = gi[n * 384 + 256 + d], hn = gh[n * 384 + 256 + d];
  const float rr = sigmoidf(ir + hr);
  const float zz = sigmoidf(iz + hz);
  const float nn = tanhf(in2 + rr * hn);
  out[t] = (1.f - zz) * nn + zz * hprev[t];
}
__global__ __launch_bounds__(256)
void final_spot_k(const float* __restrict__ gout, const float* __restrict__ gru,
                  const float* __restrict__ oW, const float* __restrict__ ob,
                  float* __restrict__ xout, float* __restrict__ oout)
{
  const long long n = ((long long)blockIdx.x * 256 + threadIdx.x) >> 5;
  const int lane = threadIdx.x & 31;
  if (n >= cN) return;
  float acc = 0.f;
#pragma unroll
  for (int j = 0; j < 4; ++j) {
    const int c = lane + j * 32;
    const float v = fmaxf(0.f, 0.5f * (gout[n * 128 + c] + gru[n * 128 + c]));
    xout[n * 128 + c] = v;
    acc += v * oW[c];
  }
  acc = wred(acc);
  if (lane == 0) oout[n] = acc + ob[0];
}
__global__ __launch_bounds__(256)
void final_word_k(const float* __restrict__ gru, const float* __restrict__ oW,
                  const float* __restrict__ ob, float* __restrict__ xout,
                  float* __restrict__ oout)
{
  const long long n = ((long long)blockIdx.x * 256 + threadIdx.x) >> 5;
  const int lane = threadIdx.x & 31;
  if (n >= cW) return;
  float acc = 0.f;
#pragma unroll
  for (int j = 0; j < 4; ++j) {
    const int c = lane + j * 32;
    const float v = fmaxf(0.f, gru[n * 128 + c]);
    xout[n * 128 + c] = v;
    acc += v * oW[c];
  }
  acc = wred(acc);
  if (lane == 0) oout[n] = acc + ob[0];
}

// ---------------------------------------------------------------------------
static inline unsigned cdiv(long long a, long long b) { return (unsigned)((a + b - 1) / b); }
static constexpr size_t ALGN(size_t x) { return (x + 255) & ~(size_t)255; }

extern "C" void kernel_launch(void* const* d_in, const int* in_sizes, int n_in,
                              void* d_out, int out_size, void* d_ws, size_t ws_size,
                              hipStream_t stream)
{
  (void)in_sizes; (void)n_in; (void)out_size; (void)ws_size;
  // ---- inputs ----
  const float* x_spot   = (const float*)d_in[0];
  const float* x_word   = (const float*)d_in[1];
  const float* aerial   = (const float*)d_in[2];
  const int*   edge_ss  = (const int*)d_in[3];
  const int*   v0 = edge_ss; const int* v1 = edge_ss + cE;
  const int*   dist     = (const int*)d_in[4];
  const int*   ws_src   = (const int*)d_in[5];
  const int*   ws_dst   = (const int*)d_in[6];
  const int*   sw_src   = (const int*)d_in[7];
  const int*   sw_dst   = (const int*)d_in[8];
  const float* att_W    = (const float*)d_in[9];
  const float* att_q    = (const float*)d_in[10];
  const float* l_ss_src = (const float*)d_in[11];
  const float* l_ss_tgt = (const float*)d_in[12];
  const float* l_ws_src = (const float*)d_in[13];
  const float* l_ws_tgt = (const float*)d_in[14];
  const float* l_sw_src = (const float*)d_in[15];
  const float* l_sw_tgt = (const float*)d_in[16];
  const float* ws_Wih   = (const float*)d_in[17];
  const float* ws_Whh   = (const float*)d_in[18];
  const float* ws_bih   = (const float*)d_in[19];
  const float* ws_bhh   = (const float*)d_in[20];
  const float* sw_Wih   = (const float*)d_in[21];
  const float* sw_Whh   = (const float*)d_in[22];
  const float* sw_bih   = (const float*)d_in[23];
  const float* sw_bhh   = (const float*)d_in[24];
  const float* dist_emb = (const float*)d_in[25];
  const float* query_W  = (const float*)d_in[26];
  const float* query_b  = (const float*)d_in[27];
  const float* value_W  = (const float*)d_in[28];
  const float* value_b  = (const float*)d_in[29];
  const float* gat_W    = (const float*)d_in[30];
  const float* gat_as   = (const float*)d_in[31];
  const float* gat_ad   = (const float*)d_in[32];
  const float* gat_b    = (const float*)d_in[33];
  const float* o_sW     = (const float*)d_in[34];
  const float* o_sb     = (const float*)d_in[35];
  const float* o_wW     = (const float*)d_in[36];
  const float* o_wb     = (const float*)d_in[37];

  // ---- workspace layout (persistent + phase-aliased scratch) ----
  char* ws = (char*)d_ws;
  size_t o = 0;
  const size_t off_spot_in = o; o += ALGN(cN * 512 * 2);   // bf16
  const size_t off_src_all = o; o += ALGN(cN * 128 * 2);   // bf16
  const size_t off_hgat    = o; o += ALGN(cN * 128 * 4);
  const size_t off_gatout  = o; o += ALGN(cN * 128 * 4);
  const size_t off_aggws   = o; o += ALGN(cN * 128 * 4);
  const size_t off_spottgt = o; o += ALGN(cN * 128 * 2);   // bf16
  const size_t off_spotgru = o; o += ALGN(cN * 128 * 4);
  const size_t off_swfeat  = o; o += ALGN(cN * 128 * 4);
  const size_t off_wsrc    = o; o += ALGN(cW * 128 * 4);
  const size_t off_aggsw   = o; o += ALGN(cW * 128 * 4);
  const size_t off_wordtgt = o; o += ALGN(cW * 128 * 2);   // bf16
  const size_t off_wordgru = o; o += ALGN(cW * 128 * 4);
  const size_t off_selsrc  = o; o += ALGN(cME * 4);
  const size_t off_seldst  = o; o += ALGN(cME * 4);
  const size_t off_keep    = o; o += ALGN(cME * 4);
  const size_t off_asad    = o; o += ALGN(cN * 2 * 4);
  const size_t off_m       = o; o += ALGN(cN * 4);
  const size_t off_wsum    = o; o += ALGN(cN * 4);
  const size_t off_cntws   = o; o += ALGN(cN * 4);
  const size_t off_cntsw   = o; o += ALGN(cW * 4);
  const size_t off_ebuf    = o; o += ALGN(cME * 4);
  const size_t off_wbuf    = o; o += ALGN(cME * 4);
  const size_t off_wq      = o; o += ALGN(4 * 512 * 4);
  const size_t off_state   = o; o += 256;
  const size_t off_scr     = o;
  // phase alpha: bf16 per-head combined attention input (4 * N * 512 * 2B)
  const size_t off_xbar    = off_scr;
  // phase beta: edge scoring buffers (dead before phase delta reuses region)
  const size_t off_tgt     = off_scr;                      // bf16 [N,128]
  const size_t off_qnode   = off_tgt   + ALGN(cN * 128 * 2);
  const size_t off_vnode   = off_qnode + ALGN(cN * 128 * 4);
  const size_t off_qdist   = off_vnode + ALGN(cN * 128 * 4);
  const size_t off_keyu    = off_qdist + ALGN(cND * 128 * 4);
  // phase delta/epsilon: GRU gate buffers (N*384 each; word pass reuses)
  const size_t off_gi      = off_scr;
  const size_t off_gh      = off_gi + ALGN(cN * 384 * 4);

  unsigned short* spot_in = (unsigned short*)(ws + off_spot_in);
  unsigned short* src_all = (unsigned short*)(ws + off_src_all);
  unsigned* st   = (unsigned*)(ws + off_state);
  float* out_xs  = (float*)d_out;                       // [N,128]
  float* out_xw  = out_xs + cN * 128;                   // [W,128]
  float* out_os  = out_xw + cW * 128;                   // [N]
  float* out_ow  = out_os + cN;                         // [W]

  // ---- 1. attention (factored) ----
  wq_k<<<cdiv(2048, 256), 256, 0, stream>>>(att_W, att_q, (float*)(ws + off_wq));
  attn_combine_k<<<cdiv(cN * 4 * 32, 256), 256, 0, stream>>>(
      x_spot, (float*)(ws + off_wq), (unsigned short*)(ws + off_xbar));
  for (int h = 0; h < 4; ++h)
    gemm<true, false, false, false, true>(stream,
        (const unsigned short*)(ws + off_xbar) + (long long)h * cN * 512,
        att_W + (long long)h * 512 * 128, nullptr,
        spot_in + h * 128, (int)cN, 128, 512, 512, 128, 512);

  // ---- 2. node projections for spot-spot branch ----
  gemm<true, false, false, false, true >(stream, spot_in, l_ss_src, nullptr, src_all, (int)cN, 128, 512, 512, 128, 128);
  gemm<true, false, false, false, true >(stream, spot_in, l_ss_tgt, nullptr, ws + off_tgt, (int)cN, 128, 512, 512, 128, 128);
  gemm<true, false, false, true , false>(stream, src_all, query_W, query_b, ws + off_qnode, (int)cN, 128, 128, 128, 128, 128);
  gemm<false, false, true , false, false>(stream, aerial, query_W + 128 * 128, nullptr, ws + off_qnode, (int)cN, 128, 512, 512, 128, 128);
  gemm<false, false, false, false, false>(stream, dist_emb, query_W + 640 * 128, nullptr, ws + off_qdist, (int)cND, 128, 128, 128, 128, 128);
  gemm<true, false, false, true , false>(stream, ws + off_tgt, value_W, value_b, ws + off_vnode, (int)cN, 128, 128, 128, 128, 128);

  // ---- 3. edge scoring + Gumbel top-K via 32-step binary search ----
  edge_logit_k<<<cdiv(cE * 32, 256), 256, 0, stream>>>(
      v0, v1, dist, (float*)(ws + off_qnode), (float*)(ws + off_qdist),
      (float*)(ws + off_vnode), (unsigned*)(ws + off_keyu));
  bisect_init_k<<<1, 32, 0, stream>>>(st);
  for (int it = 0; it < 32; ++it) {
    count_ge_k<<<cdiv(cE, 256), 256, 0, stream>>>((unsigned*)(ws + off_keyu), st);
    bisect_step_k<<<1, 32, 0, stream>>>(st);
  }
  select_gt_k<<<cdiv(cE, 256), 256, 0, stream>>>((unsigned*)(ws + off_keyu), v0, v1, st,
      (int*)(ws + off_selsrc), (int*)(ws + off_seldst), (int*)(ws + off_keep));
  select_eq_k<<<cdiv(cE, 256), 256, 0, stream>>>((unsigned*)(ws + off_keyu), v0, v1, st,
      (int*)(ws + off_selsrc), (int*)(ws + off_seldst), (int*)(ws + off_keep));

  // ---- 4. GAT over sampled edges + self loops ----
  gat_init_k<<<cdiv(cN * 128, 256), 256, 0, stream>>>(gat_b, (float*)(ws + off_gatout),
      (float*)(ws + off_m), (float*)(ws + off_wsum),
      (int*)(ws + off_selsrc), (int*)(ws + off_seldst), (int*)(ws + off_keep));
  gemm<true, false, false, false, false>(stream, src_all, gat_W, nullptr, ws + off_hgat, (int)cN, 128, 128, 128, 128, 128);
  asad_k<<<cdiv(cN * 32, 256), 256, 0, stream>>>((float*)(ws + off_hgat), gat_as, gat_ad, (float*)(ws + off_asad));
  gat_e_k<<<cdiv(cME, 256), 256, 0, stream>>>((int*)(ws + off_selsrc), (int*)(ws + off_seldst),
      (int*)(ws + off_keep), (float*)(ws + off_asad), (float*)(ws + off_ebuf), (float*)(ws + off_m));
  gat_w_k<<<cdiv(cME, 256), 256, 0, stream>>>((int*)(ws + off_seldst), (float*)(ws + off_ebuf),
      (float*)(ws + off_m), (float*)(ws + off_wbuf), (float*)(ws + off_wsum));
  gat_scatter_k<<<cdiv(cME * 32, 256), 256, 0, stream>>>((int*)(ws + off_selsrc), (int*)(ws + off_seldst),
      (float*)(ws + off_wbuf), (float*)(ws + off_wsum), (float*)(ws + off_hgat), (float*)(ws + off_gatout));

  // ---- 5. word -> spot branch ----
  gemm<false, false, false, false, false>(stream, x_word, l_ws_src, nullptr, ws + off_wsrc, (int)cW, 128, 300, 300, 128, 128);
  fill_k<<<cdiv(cN * 128, 256), 256, 0, stream>>>((float*)(ws + off_aggws), 0.f, cN * 128);
  fill_k<<<cdiv(cN, 256), 256, 0, stream>>>((float*)(ws + off_cntws), 0.f, cN);
  scatter_mean_k<<<cdiv(cEW * 32, 256), 256, 0, stream>>>(ws_src, ws_dst, (float*)(ws + off_wsrc),
      (float*)(ws + off_aggws), (float*)(ws + off_cntws), cEW);
  div_cnt_k<<<cdiv(cN * 128, 256), 256, 0, stream>>>((float*)(ws + off_aggws), (float*)(ws + off_cntws), cN);
  gemm<true, false, false, false, true >(stream, spot_in, l_ws_tgt, nullptr, ws + off_spottgt, (int)cN, 128, 512, 512, 128, 128);
  gemm<true, true , false, true , false>(stream, ws + off_spottgt, ws_Wih, ws_bih, ws + off_gi, (int)cN, 384, 128, 128, 128, 384);
  gemm<false, true , false, true , false>(stream, ws + off_aggws, ws_Whh, ws_bhh, ws + off_gh, (int)cN, 384, 128, 128, 128, 384);
  gru_k<<<cdiv(cN * 128, 256), 256, 0, stream>>>((float*)(ws + off_gi), (float*)(ws + off_gh),
      (float*)(ws + off_aggws), (float*)(ws + off_spotgru), cN);

  // ---- 6. spot -> word branch ----
  gemm<true, false, false, false, false>(stream, spot_in, l_sw_src, nullptr, ws + off_swfeat, (int)cN, 128, 512, 512, 128, 128);
  fill_k<<<cdiv(cW * 128, 256), 256, 0, stream>>>((float*)(ws + off_aggsw), 0.f, cW * 128);
  fill_k<<<cdiv(cW, 256), 256, 0, stream>>>((float*)(ws + off_cntsw), 0.f, cW);
  scatter_mean_k<<<cdiv(cESW * 32, 256), 256, 0, stream>>>(sw_src, sw_dst, (float*)(ws + off_swfeat),
      (float*)(ws + off_aggsw), (float*)(ws + off_cntsw), cESW);
  div_cnt_k<<<cdiv(cW * 128, 256), 256, 0, stream>>>((float*)(ws + off_aggsw), (float*)(ws + off_cntsw), cW);
  gemm<false, false, false, false, true >(stream, x_word, l_sw_tgt, nullptr, ws + off_wordtgt, (int)cW, 128, 300, 300, 128, 128);
  gemm<true, true , false, true , false>(stream, ws + off_wordtgt, sw_Wih, sw_bih, ws + off_gi, (int)cW, 384, 128, 128, 128, 384);
  gemm<false, true , false, true , false>(stream, ws + off_aggsw, sw_Whh, sw_bhh, ws + off_gh, (int)cW, 384, 128, 128, 128, 384);
  gru_k<<<cdiv(cW * 128, 256), 256, 0, stream>>>((float*)(ws + off_gi), (float*)(ws + off_gh),
      (float*)(ws + off_aggsw), (float*)(ws + off_wordgru), cW);

  // ---- 7. outputs ----
  final_spot_k<<<cdiv(cN * 32, 256), 256, 0, stream>>>((float*)(ws + off_gatout),
      (float*)(ws + off_spotgru), o_sW, o_sb, out_xs, out_os);
  final_word_k<<<cdiv(cW * 32, 256), 256, 0, stream>>>((float*)(ws + off_wordgru),
      o_wW, o_wb, out_xw, out_ow);
}
